// VerificationRewardLoss_12747462935019
// MI455X (gfx1250) — matrix operand order repaired and checked by reading, not compiled, not run on hardware
//
#include <hip/hip_runtime.h>
#include <stdint.h>

// ---------------------------------------------------------------------------
// VerificationRewardLoss for MI455X (gfx1250, wave32, WMMA).
//
// loss = -0.5*(CE(p2) - CE(p3)) + 0.1*mean(cos(p2,p3))
// CE needs only per-row logsumexp + target logit => fused GEMM + online softmax.
// GEMM path: split-bf16 (hi/lo) with v_wmma_f32_16x16x32_bf16, fp32 accum,
// ping-pong LDS double buffering with register prefetch of the next K-chunk.
// Wave layout 2(M)x4(N): each wave computes 2 M-tiles x 2 N-subtiles so both
// A and B fragments are reused twice (16 ds_load_b128 per 12 WMMAs).
// ---------------------------------------------------------------------------

typedef __attribute__((ext_vector_type(16))) __bf16 v16bf;
typedef __attribute__((ext_vector_type(8)))  __bf16 v8bf;
typedef __attribute__((ext_vector_type(8)))  float  v8f;

#define D_DIM      768
#define V_DIM      50257
#define ROWS_PASS  4096          // B*T
#define ROWS_ALL   8192          // both passes concatenated
#define NCHUNK     32            // vocab chunks (grid.y)
#define NTILE      393           // ceil(V / 128)
#define BM         64            // rows per block
#define BN         128           // vocab cols per tile iteration
#define KC         32            // K per WMMA step
#define KSTEPS     (D_DIM / KC)  // 24
#define AST        40            // padded LDS row stride (bf16 elems) for A
#define BST        40            // padded LDS row stride (bf16 elems) for B

__device__ __forceinline__ void split_store(__bf16* hi, __bf16* lo, int idx, float x) {
  __bf16 h = (__bf16)x;                 // RNE to bf16
  hi[idx] = h;
  lo[idx] = (__bf16)(x - (float)h);     // residual
}

// ---------------------------------------------------------------------------
// Kernel 1: fused logits GEMM + online softmax partials.
// grid = (ROWS_ALL/BM, NCHUNK), block = 256 (8 waves: 2 along M x 4 along N)
// ---------------------------------------------------------------------------
__global__ __launch_bounds__(256) void ce_gemm_kernel(
    const float* __restrict__ p2, const float* __restrict__ p3,
    const float* __restrict__ W,
    float* __restrict__ partM, float* __restrict__ partS)
{
  __shared__ __bf16 AhiB[2][BM * AST];
  __shared__ __bf16 AloB[2][BM * AST];
  __shared__ __bf16 BhiB[2][BN * BST];
  __shared__ __bf16 BloB[2][BN * BST];
  __shared__ float  pm[4][BM], ps[4][BM];
  __shared__ float  rowM[BM], rowS[BM];

  const int tid   = threadIdx.x;
  const int lane  = tid & 31;
  const int wave  = tid >> 5;        // 0..7
  const int waveM = wave >> 2;       // 0..1 : 32-row slice (2 M-tiles)
  const int waveN = wave & 3;        // 0..3 : 32-col slice (2 N-subtiles)
  const int half  = lane >> 4;       // wave32 half
  const int lr    = lane & 15;

  const int gRow0 = blockIdx.x * BM;                   // [0, 8192)
  const float* H  = (gRow0 < ROWS_PASS) ? p2 : p3;
  const int hRow0 = gRow0 & (ROWS_PASS - 1);

  if (tid < BM) { rowM[tid] = -3.0e38f; rowS[tid] = 0.0f; }

  for (int t = blockIdx.y; t < NTILE; t += NCHUNK) {
    const int v0 = t * BN;

    v8f acc[2][2];
    #pragma unroll
    for (int mt = 0; mt < 2; ++mt)
      #pragma unroll
      for (int s = 0; s < 2; ++s)
        #pragma unroll
        for (int j = 0; j < 8; ++j) acc[mt][s][j] = 0.0f;

    // ---- prologue: fetch chunk 0 to registers, stage into buffer 0 ----
    float4 aR[2], bR[4];
    #pragma unroll
    for (int i = 0; i < 2; ++i) {
      int q = tid * 2 + i, r = q >> 3, kq = (q & 7) << 2;
      aR[i] = *(const float4*)&H[(size_t)(hRow0 + r) * D_DIM + kq];
    }
    #pragma unroll
    for (int i = 0; i < 4; ++i) {
      int q = tid * 4 + i, r = q >> 3, kq = (q & 7) << 2;
      int vv = v0 + r;
      bR[i] = (vv < V_DIM) ? *(const float4*)&W[(size_t)vv * D_DIM + kq]
                           : make_float4(0.f, 0.f, 0.f, 0.f);
    }
    #pragma unroll
    for (int i = 0; i < 2; ++i) {
      int q = tid * 2 + i, r = q >> 3, kq = (q & 7) << 2;
      split_store(AhiB[0], AloB[0], r * AST + kq + 0, aR[i].x);
      split_store(AhiB[0], AloB[0], r * AST + kq + 1, aR[i].y);
      split_store(AhiB[0], AloB[0], r * AST + kq + 2, aR[i].z);
      split_store(AhiB[0], AloB[0], r * AST + kq + 3, aR[i].w);
    }
    #pragma unroll
    for (int i = 0; i < 4; ++i) {
      int q = tid * 4 + i, r = q >> 3, kq = (q & 7) << 2;
      split_store(BhiB[0], BloB[0], r * BST + kq + 0, bR[i].x);
      split_store(BhiB[0], BloB[0], r * BST + kq + 1, bR[i].y);
      split_store(BhiB[0], BloB[0], r * BST + kq + 2, bR[i].z);
      split_store(BhiB[0], BloB[0], r * BST + kq + 3, bR[i].w);
    }
    __syncthreads();

    for (int ks = 0; ks < KSTEPS; ++ks) {
      const int cur  = ks & 1;
      const int nxt  = cur ^ 1;
      const bool more = (ks + 1) < KSTEPS;
      const int kn = (ks + 1) * KC;

      // ---- issue next chunk's global loads early (latency hides under WMMA)
      if (more) {
        #pragma unroll
        for (int i = 0; i < 2; ++i) {
          int q = tid * 2 + i, r = q >> 3, kq = (q & 7) << 2;
          aR[i] = *(const float4*)&H[(size_t)(hRow0 + r) * D_DIM + kn + kq];
        }
        #pragma unroll
        for (int i = 0; i < 4; ++i) {
          int q = tid * 4 + i, r = q >> 3, kq = (q & 7) << 2;
          int vv = v0 + r;
          bR[i] = (vv < V_DIM) ? *(const float4*)&W[(size_t)vv * D_DIM + kn + kq]
                               : make_float4(0.f, 0.f, 0.f, 0.f);
        }
      }

      // ---- A fragments (2 M-tiles), 16x32 bf16 layout:
      // lanes0-15: K=0..7 | 16..23 ; lanes16-31: K=8..15 | 24..31
      v16bf ah[2], al[2];
      #pragma unroll
      for (int mt = 0; mt < 2; ++mt) {
        const int arow = waveM * 32 + mt * 16 + lr;
        v8bf h0 = *(const v8bf*)&AhiB[cur][arow * AST + half * 8];
        v8bf h1 = *(const v8bf*)&AhiB[cur][arow * AST + 16 + half * 8];
        v8bf l0 = *(const v8bf*)&AloB[cur][arow * AST + half * 8];
        v8bf l1 = *(const v8bf*)&AloB[cur][arow * AST + 16 + half * 8];
        #pragma unroll
        for (int i = 0; i < 8; ++i) {
          ah[mt][i] = h0[i]; ah[mt][8 + i] = h1[i];
          al[mt][i] = l0[i]; al[mt][8 + i] = l1[i];
        }
      }

      // ---- B fragments (2 N-subtiles) + 3 WMMAs each per M-tile
      #pragma unroll
      for (int s = 0; s < 2; ++s) {
        const int brow = waveN * 32 + s * 16 + lr;   // column n of B (K x N)
        v8bf h0 = *(const v8bf*)&BhiB[cur][brow * BST + half * 8];
        v8bf h1 = *(const v8bf*)&BhiB[cur][brow * BST + 16 + half * 8];
        v8bf l0 = *(const v8bf*)&BloB[cur][brow * BST + half * 8];
        v8bf l1 = *(const v8bf*)&BloB[cur][brow * BST + 16 + half * 8];
        v16bf bh, bl;
        #pragma unroll
        for (int i = 0; i < 8; ++i) {
          bh[i] = h0[i]; bh[8 + i] = h1[i];
          bl[i] = l0[i]; bl[8 + i] = l1[i];
        }
        #pragma unroll
        for (int mt = 0; mt < 2; ++mt) {
          acc[mt][s] = __builtin_amdgcn_wmma_f32_16x16x32_bf16(
                         false, ah[mt], false, bh, (short)0, acc[mt][s], false, false);
          acc[mt][s] = __builtin_amdgcn_wmma_f32_16x16x32_bf16(
                         false, ah[mt], false, bl, (short)0, acc[mt][s], false, false);
          acc[mt][s] = __builtin_amdgcn_wmma_f32_16x16x32_bf16(
                         false, al[mt], false, bh, (short)0, acc[mt][s], false, false);
        }
      }

      // ---- stage prefetched chunk into the other buffer ----
      if (more) {
        #pragma unroll
        for (int i = 0; i < 2; ++i) {
          int q = tid * 2 + i, r = q >> 3, kq = (q & 7) << 2;
          split_store(AhiB[nxt], AloB[nxt], r * AST + kq + 0, aR[i].x);
          split_store(AhiB[nxt], AloB[nxt], r * AST + kq + 1, aR[i].y);
          split_store(AhiB[nxt], AloB[nxt], r * AST + kq + 2, aR[i].z);
          split_store(AhiB[nxt], AloB[nxt], r * AST + kq + 3, aR[i].w);
        }
        #pragma unroll
        for (int i = 0; i < 4; ++i) {
          int q = tid * 4 + i, r = q >> 3, kq = (q & 7) << 2;
          split_store(BhiB[nxt], BloB[nxt], r * BST + kq + 0, bR[i].x);
          split_store(BhiB[nxt], BloB[nxt], r * BST + kq + 1, bR[i].y);
          split_store(BhiB[nxt], BloB[nxt], r * BST + kq + 2, bR[i].z);
          split_store(BhiB[nxt], BloB[nxt], r * BST + kq + 3, bR[i].w);
        }
      }
      __syncthreads();
    } // k loop

    // ---- per-row (max, sum-exp) over this wave's 32 columns ----
    // D layout: lane n = lr, VGPR j -> row = j + 8*half
    float m8[2][8], p8[2][8];
    #pragma unroll
    for (int mt = 0; mt < 2; ++mt) {
      #pragma unroll
      for (int j = 0; j < 8; ++j) {
        float x[2];
        #pragma unroll
        for (int s = 0; s < 2; ++s) {
          int vv = v0 + waveN * 32 + s * 16 + lr;
          x[s] = (vv < V_DIM) ? acc[mt][s][j] : -3.0e38f;
        }
        float m = fmaxf(x[0], x[1]);
        #pragma unroll
        for (int off = 1; off <= 8; off <<= 1)
          m = fmaxf(m, __shfl_xor(m, off, 32));     // within 16-lane half
        float p = __expf(x[0] - m) + __expf(x[1] - m);
        #pragma unroll
        for (int off = 1; off <= 8; off <<= 1)
          p += __shfl_xor(p, off, 32);
        m8[mt][j] = m; p8[mt][j] = p;
      }
    }

    if (lr == 0) {
      #pragma unroll
      for (int mt = 0; mt < 2; ++mt)
        #pragma unroll
        for (int j = 0; j < 8; ++j) {
          int r = waveM * 32 + mt * 16 + half * 8 + j;
          pm[waveN][r] = m8[mt][j];
          ps[waveN][r] = p8[mt][j];
        }
    }
    __syncthreads();

    // merge the four N-waves, then fold into the running per-row state
    if (tid < BM) {
      float M = fmaxf(fmaxf(pm[0][tid], pm[1][tid]), fmaxf(pm[2][tid], pm[3][tid]));
      float S = __expf(pm[0][tid] - M) * ps[0][tid]
              + __expf(pm[1][tid] - M) * ps[1][tid]
              + __expf(pm[2][tid] - M) * ps[2][tid]
              + __expf(pm[3][tid] - M) * ps[3][tid];
      float Mr = rowM[tid], Sr = rowS[tid];
      float Mn = fmaxf(Mr, M);
      rowS[tid] = Sr * __expf(Mr - Mn) + S * __expf(M - Mn);
      rowM[tid] = Mn;
    }
  } // v-tile loop

  if (tid < BM) {
    int gr = gRow0 + tid;
    partM[(size_t)gr * NCHUNK + blockIdx.y] = rowM[tid];
    partS[(size_t)gr * NCHUNK + blockIdx.y] = rowS[tid];
  }
}

// ---------------------------------------------------------------------------
// Kernel 2: per-row target logit dot(H_row, W[target])  (fp32, exact)
// ---------------------------------------------------------------------------
__global__ __launch_bounds__(256) void tgt_kernel(
    const float* __restrict__ p2, const float* __restrict__ p3,
    const float* __restrict__ W, const int* __restrict__ tg,
    float* __restrict__ tout)
{
  const int row = blockIdx.x;                       // 0..8191
  const float* H = (row < ROWS_PASS) ? p2 : p3;
  const int hr = row & (ROWS_PASS - 1);
  const int t  = tg[hr];
  float s = 0.0f;
  if (t >= 0) {
    const float* h = H + (size_t)hr * D_DIM;
    const float* w = W + (size_t)t * D_DIM;
    for (int d = threadIdx.x; d < D_DIM; d += 256) s += h[d] * w[d];
  }
  #pragma unroll
  for (int off = 1; off <= 16; off <<= 1) s += __shfl_xor(s, off, 32);
  __shared__ float red[8];
  if ((threadIdx.x & 31) == 0) red[threadIdx.x >> 5] = s;
  __syncthreads();
  if (threadIdx.x == 0) {
    float tot = 0.0f;
    #pragma unroll
    for (int i = 0; i < 8; ++i) tot += red[i];
    tout[row] = tot;
  }
}

// ---------------------------------------------------------------------------
// Kernel 3: per-row cosine similarity (torch eps semantics)
// ---------------------------------------------------------------------------
__global__ __launch_bounds__(256) void cos_kernel(
    const float* __restrict__ p2, const float* __restrict__ p3,
    float* __restrict__ cosv)
{
  const int row = blockIdx.x;                       // 0..4095
  const float* a = p2 + (size_t)row * D_DIM;
  const float* b = p3 + (size_t)row * D_DIM;
  float aa = 0.f, bb = 0.f, ab = 0.f;
  for (int d = threadIdx.x; d < D_DIM; d += 256) {
    float x = a[d], y = b[d];
    aa += x * x; bb += y * y; ab += x * y;
  }
  #pragma unroll
  for (int off = 1; off <= 16; off <<= 1) {
    aa += __shfl_xor(aa, off, 32);
    bb += __shfl_xor(bb, off, 32);
    ab += __shfl_xor(ab, off, 32);
  }
  __shared__ float ra[8], rb[8], rc[8];
  if ((threadIdx.x & 31) == 0) {
    int w = threadIdx.x >> 5;
    ra[w] = aa; rb[w] = bb; rc[w] = ab;
  }
  __syncthreads();
  if (threadIdx.x == 0) {
    float A = 0.f, Bv = 0.f, C = 0.f;
    #pragma unroll
    for (int i = 0; i < 8; ++i) { A += ra[i]; Bv += rb[i]; C += rc[i]; }
    float na = fmaxf(sqrtf(A), 1e-8f);
    float nb = fmaxf(sqrtf(Bv), 1e-8f);
    cosv[row] = C / (na * nb);
  }
}

// ---------------------------------------------------------------------------
// Kernel 4: combine everything into the scalar loss (single block)
// ---------------------------------------------------------------------------
__global__ __launch_bounds__(256) void final_kernel(
    const float* __restrict__ partM, const float* __restrict__ partS,
    const float* __restrict__ tout,  const float* __restrict__ cosv,
    const int* __restrict__ tg, float* __restrict__ out)
{
  float s2 = 0.f, s3 = 0.f, cs = 0.f; int cnt = 0;
  for (int r = threadIdx.x; r < ROWS_ALL; r += 256) {
    int t = tg[r & (ROWS_PASS - 1)];
    if (t < 0) continue;
    const float* pmr = partM + (size_t)r * NCHUNK;
    const float* psr = partS + (size_t)r * NCHUNK;
    float M = -3.0e38f;
    for (int c = 0; c < NCHUNK; ++c) M = fmaxf(M, pmr[c]);
    float S = 0.f;
    for (int c = 0; c < NCHUNK; ++c) S += expf(pmr[c] - M) * psr[c];
    float nll = M + logf(S) - tout[r];
    if (r < ROWS_PASS) { s2 += nll; cnt++; } else { s3 += nll; }
  }
  for (int r = threadIdx.x; r < ROWS_PASS; r += 256) cs += cosv[r];

  #pragma unroll
  for (int off = 1; off <= 16; off <<= 1) {
    s2 += __shfl_xor(s2, off, 32);
    s3 += __shfl_xor(s3, off, 32);
    cs += __shfl_xor(cs, off, 32);
    cnt += __shfl_xor(cnt, off, 32);
  }
  __shared__ float r2[8], r3[8], rcs[8]; __shared__ int rcn[8];
  if ((threadIdx.x & 31) == 0) {
    int w = threadIdx.x >> 5;
    r2[w] = s2; r3[w] = s3; rcs[w] = cs; rcn[w] = cnt;
  }
  __syncthreads();
  if (threadIdx.x == 0) {
    float S2 = 0.f, S3 = 0.f, CS = 0.f; int CN = 0;
    #pragma unroll
    for (int i = 0; i < 8; ++i) { S2 += r2[i]; S3 += r3[i]; CS += rcs[i]; CN += rcn[i]; }
    float denom = fmaxf((float)CN, 1.0f);
    float ce2 = S2 / denom, ce3 = S3 / denom;
    out[0] = -0.5f * (ce2 - ce3) + 0.1f * (CS / (float)ROWS_PASS);
  }
}

// ---------------------------------------------------------------------------
extern "C" void kernel_launch(void* const* d_in, const int* in_sizes, int n_in,
                              void* d_out, int out_size, void* d_ws, size_t ws_size,
                              hipStream_t stream) {
  const float* p2 = (const float*)d_in[0];
  const float* p3 = (const float*)d_in[1];
  const float* W  = (const float*)d_in[2];
  const int*   tg = (const int*)d_in[3];

  float* ws    = (float*)d_ws;
  float* partM = ws;                                        // 8192*32
  float* partS = partM + (size_t)ROWS_ALL * NCHUNK;         // 8192*32
  float* tout  = partS + (size_t)ROWS_ALL * NCHUNK;         // 8192
  float* cosv  = tout + ROWS_ALL;                           // 4096
  // total ws use: ~2.1 MB

  dim3 g1(ROWS_ALL / BM, NCHUNK);
  ce_gemm_kernel<<<g1, 256, 0, stream>>>(p2, p3, W, partM, partS);
  tgt_kernel<<<ROWS_ALL, 256, 0, stream>>>(p2, p3, W, tg, tout);
  cos_kernel<<<ROWS_PASS, 256, 0, stream>>>(p2, p3, cosv);
  final_kernel<<<1, 256, 0, stream>>>(partM, partS, tout, cosv, tg, (float*)d_out);
}
